// FastTextModel_41137196761528
// MI455X (gfx1250) — matrix operand order repaired and checked by reading, compiled, not run on hardware
//
#include <hip/hip_runtime.h>
#include <math.h>

// ---------------------------------------------------------------------------
// Types for CDNA5 WMMA (wave32): v_wmma_f32_16x16x32_bf16
// ---------------------------------------------------------------------------
typedef __attribute__((ext_vector_type(16))) __bf16 bf16x16;
typedef __attribute__((ext_vector_type(8)))  __bf16 bf16x8;
typedef __attribute__((ext_vector_type(4)))  __bf16 bf16x4;
typedef __attribute__((ext_vector_type(8)))  float  f32x8;
typedef __attribute__((ext_vector_type(4)))  float  f32x4;

#define L2E 1.4426950408889634f
#define LN2 0.6931471805599453f
#define NEG_BIG (-1.0e30f)

static constexpr int BATCH     = 2048;
static constexpr int EMBED     = 128;
static constexpr int SEQ       = 50;
static constexpr int CUT0      = 12500;           // NUM_CLASSES/4
static constexpr int HEAD_N    = CUT0 + 2;        // 12502
static constexpr int T0_N      = 12500;           // CUT1 - CUT0
static constexpr int T1_N      = 25000;           // NUM_CLASSES - CUT1
static constexpr int OUT_N     = 50000;
static constexpr int H0DIM     = 64;
static constexpr int H1DIM     = 32;

// ---------------------------------------------------------------------------
// Kernel 1: embedding-bag mean pool (f32) + fused H0/H1 projections.
// 1 wave per row (32 lanes x float4 = 128 features), 8 rows per block.
// ---------------------------------------------------------------------------
__global__ __launch_bounds__(256) void pool_proj_kernel(
    const int*   __restrict__ x,        // (2048, 50)
    const float* __restrict__ emb,      // (1e6, 128)
    const float* __restrict__ t0_proj,  // (64, 128)
    const float* __restrict__ t1_proj,  // (32, 128)
    __bf16*      __restrict__ pooled_b, // (2048, 128) bf16
    __bf16*      __restrict__ h0_b,     // (2048, 64)  bf16
    __bf16*      __restrict__ h1_b)     // (2048, 32)  bf16
{
    __shared__ float srow[8][EMBED];

    const int wave = threadIdx.x >> 5;
    const int lane = threadIdx.x & 31;
    const int row  = blockIdx.x * 8 + wave;

    const int* xr = x + (size_t)row * SEQ;
    float ax = 0.f, ay = 0.f, az = 0.f, aw = 0.f;
    int cnt = 0;
    for (int t = 0; t < SEQ; ++t) {
        const int idx = xr[t];
        if (idx != 0) {
            ++cnt;
            // streaming gather from the 512MB table: keep it out of L2's way
            const f32x4* ep = (const f32x4*)(emb + (size_t)idx * EMBED) + lane;
            const f32x4 e = __builtin_nontemporal_load(ep);
            ax += e.x; ay += e.y; az += e.z; aw += e.w;
        }
    }
    const float inv = 1.0f / (float)max(cnt, 1);
    ax *= inv; ay *= inv; az *= inv; aw *= inv;

    // bf16 pooled for the WMMA A-matrix
    bf16x4 pb = { (__bf16)ax, (__bf16)ay, (__bf16)az, (__bf16)aw };
    *(bf16x4*)(pooled_b + (size_t)row * EMBED + lane * 4) = pb;

    srow[wave][lane * 4 + 0] = ax;
    srow[wave][lane * 4 + 1] = ay;
    srow[wave][lane * 4 + 2] = az;
    srow[wave][lane * 4 + 3] = aw;
    __syncthreads();

    // h0[j] = pooled . t0_proj[j]  (j = lane, lane+32) ; h1[j] (j = lane)
    const float* p0a = t0_proj + (size_t)lane * EMBED;
    const float* p0b = t0_proj + (size_t)(lane + 32) * EMBED;
    const float* p1  = t1_proj + (size_t)lane * EMBED;
    float d0 = 0.f, d1 = 0.f, d2 = 0.f;
    #pragma unroll 4
    for (int k = 0; k < EMBED; ++k) {
        const float pv = srow[wave][k];
        d0 = fmaf(pv, p0a[k], d0);
        d1 = fmaf(pv, p0b[k], d1);
        d2 = fmaf(pv, p1[k],  d2);
    }
    h0_b[(size_t)row * H0DIM + lane]      = (__bf16)d0;
    h0_b[(size_t)row * H0DIM + lane + 32] = (__bf16)d1;
    h1_b[(size_t)row * H1DIM + lane]      = (__bf16)d2;
}

// ---------------------------------------------------------------------------
// Kernel 2: f32 -> bf16 weight conversion (vectorized x4).
// ---------------------------------------------------------------------------
__global__ __launch_bounds__(256) void cvt_bf16_kernel(
    const float* __restrict__ src, __bf16* __restrict__ dst, int n4)
{
    const int i = blockIdx.x * blockDim.x + threadIdx.x;
    if (i < n4) {
        const f32x4 v = ((const f32x4*)src)[i];
        bf16x4 o = { (__bf16)v.x, (__bf16)v.y, (__bf16)v.z, (__bf16)v.w };
        *(bf16x4*)(dst + (size_t)i * 4) = o;
    }
}

// ---------------------------------------------------------------------------
// Helpers
// ---------------------------------------------------------------------------
// Branchless combine of two (max, sumexp) pairs. Finite init => no NaN guard.
__device__ __forceinline__ void lse_combine(float& M, float& S, float m2, float s2)
{
    const float Mn = fmaxf(M, m2);
    S = S * exp2f((M - Mn) * L2E) + s2 * exp2f((m2 - Mn) * L2E);
    M = Mn;
}

// Branchless single-exp online update: if v>M: S=S*e+1 else S=S+e,
// with e = 2^(-|v-M|*log2e). Exact match of the two-case form.
__device__ __forceinline__ void lse_update(float& M, float& S, float v)
{
    const float e  = exp2f(-fabsf(v - M) * L2E);
    const bool  gt = v > M;
    S = fmaf(gt ? S : 1.0f, e, gt ? 1.0f : S);
    M = fmaxf(M, v);
}

__device__ __forceinline__ bf16x16 join8(bf16x8 lo, bf16x8 hi)
{
    return __builtin_shufflevector(lo, hi, 0,1,2,3,4,5,6,7,8,9,10,11,12,13,14,15);
}

// ---------------------------------------------------------------------------
// Kernel 3: fused GEMM + log_softmax segment kernel (recompute strategy).
//   MODE 0: head  (A = pooled bf16, K=128). Writes out[:, 0:12500] and the
//           two tail-gate log-probs to head_lp_out (2048 x 2).
//   MODE 1: tail0 (A = h0 bf16, K=64).  out[:, 12500:25000] += head_lp[:,0]
//   MODE 2: tail1 (A = h1 bf16, K=32).  out[:, 25000:50000] += head_lp[:,1]
// One block = 16 rows, 8 waves stride over N-tiles of 16 columns.
// WMMA: D(16x16,f32) += A(16x32,bf16) x B(32x16,bf16), f32 accumulate.
// ---------------------------------------------------------------------------
template<int KSTEPS, int MODE>
__global__ __launch_bounds__(256) void seg_softmax_kernel(
    const __bf16* __restrict__ A,          // (2048, K)
    const __bf16* __restrict__ W,          // (NW, K) bf16 row-major
    const float*  __restrict__ head_lp_in, // (2048, 2)
    float*        __restrict__ head_lp_out,// (2048, 2)
    float*        __restrict__ out,        // (2048, 50000)
    int Nvalid, int NW, int Nwrite, int outOff)
{
    constexpr int K = 32 * KSTEPS;

    const int lane = threadIdx.x & 31;
    const int wave = threadIdx.x >> 5;
    const int h    = lane >> 4;     // lane half
    const int nl   = lane & 15;
    const int row0 = blockIdx.x * 16;

    // ---- A fragments: lane holds row (row0+nl); per k-block, K at
    //      k = 16h + {0..15} (V0-7), matching B. Two b128 loads.
    bf16x16 afrag[KSTEPS];
    {
        const __bf16* ar = A + (size_t)(row0 + nl) * K;
        #pragma unroll
        for (int kk = 0; kk < KSTEPS; ++kk) {
            const __bf16* p = ar + kk * 32 + h * 16;
            afrag[kk] = join8(*(const bf16x8*)p, *(const bf16x8*)(p + 8));
        }
    }

    float Mr[8], Sr[8];
    #pragma unroll
    for (int r = 0; r < 8; ++r) { Mr[r] = NEG_BIG; Sr[r] = 0.0f; }

    const int numTiles = (Nvalid + 15) >> 4;

    // ================= Phase 1: logits -> per-row (max, sumexp) ============
    for (int j = wave; j < numTiles; j += 8) {
        const int n0 = j << 4;
        int nw = n0 + nl; if (nw >= NW) nw = NW - 1;          // clamp pad rows
        const __bf16* wr = W + (size_t)nw * K;

        // issue all B-fragment loads first (one clause), then the WMMA chain
        bf16x16 bfrag[KSTEPS];
        #pragma unroll
        for (int kk = 0; kk < KSTEPS; ++kk) {
            const __bf16* p = wr + kk * 32 + h * 16;
            bfrag[kk] = join8(*(const bf16x8*)p, *(const bf16x8*)(p + 8));
        }
        f32x8 acc = {};
        #pragma unroll
        for (int kk = 0; kk < KSTEPS; ++kk) {
            acc = __builtin_amdgcn_wmma_f32_16x16x32_bf16(
                false, afrag[kk], false, bfrag[kk], (short)0, acc, false, false);
        }
        const bool valid = (n0 + nl) < Nvalid;
        #pragma unroll
        for (int r = 0; r < 8; ++r) {
            const float v = valid ? acc[r] : NEG_BIG;   // folds to S += 0
            lse_update(Mr[r], Sr[r], v);
        }
    }

    // ---- reduce across the 16 lanes of each half-wave
    #pragma unroll
    for (int mask = 1; mask < 16; mask <<= 1) {
        #pragma unroll
        for (int r = 0; r < 8; ++r) {
            const float m2 = __shfl_xor(Mr[r], mask, 32);
            const float s2 = __shfl_xor(Sr[r], mask, 32);
            lse_combine(Mr[r], Sr[r], m2, s2);
        }
    }

    __shared__ float s_max[8][16];
    __shared__ float s_sum[8][16];
    __shared__ float s_lse[16];
    if (nl == 0) {
        #pragma unroll
        for (int r = 0; r < 8; ++r) {
            s_max[wave][h * 8 + r] = Mr[r];
            s_sum[wave][h * 8 + r] = Sr[r];
        }
    }
    __syncthreads();
    if (threadIdx.x < 16) {
        float M = NEG_BIG, S = 0.0f;
        #pragma unroll
        for (int w = 0; w < 8; ++w)
            lse_combine(M, S, s_max[w][threadIdx.x], s_sum[w][threadIdx.x]);
        s_lse[threadIdx.x] = M + LN2 * log2f(S);
    }
    __syncthreads();

    float lse_r[8], bias_r[8];
    #pragma unroll
    for (int r = 0; r < 8; ++r) {
        const int m = r + h * 8;
        lse_r[r]  = s_lse[m];
        bias_r[r] = (MODE == 0) ? 0.0f
                  : head_lp_in[(size_t)(row0 + m) * 2 + (MODE - 1)];
    }

    // ================= Phase 2: recompute logits, write normalized =========
    for (int j = wave; j < numTiles; j += 8) {
        const int n0 = j << 4;
        int nw = n0 + nl; if (nw >= NW) nw = NW - 1;
        const __bf16* wr = W + (size_t)nw * K;

        bf16x16 bfrag[KSTEPS];
        #pragma unroll
        for (int kk = 0; kk < KSTEPS; ++kk) {
            const __bf16* p = wr + kk * 32 + h * 16;
            bfrag[kk] = join8(*(const bf16x8*)p, *(const bf16x8*)(p + 8));
        }
        f32x8 acc = {};
        #pragma unroll
        for (int kk = 0; kk < KSTEPS; ++kk) {
            acc = __builtin_amdgcn_wmma_f32_16x16x32_bf16(
                false, afrag[kk], false, bfrag[kk], (short)0, acc, false, false);
        }
        const int n = n0 + nl;
        if (n < Nwrite) {
            // output is write-once, never re-read: stream past L2
            float* op = out + (size_t)row0 * OUT_N + outOff + n;
            #pragma unroll
            for (int r = 0; r < 8; ++r) {
                const int m = r + h * 8;
                __builtin_nontemporal_store(acc[r] - lse_r[r] + bias_r[r],
                                            op + (size_t)m * OUT_N);
            }
        } else if (MODE == 0 && n < Nvalid) {
            #pragma unroll
            for (int r = 0; r < 8; ++r) {
                const int m = r + h * 8;
                head_lp_out[(size_t)(row0 + m) * 2 + (n - Nwrite)] =
                    acc[r] - lse_r[r];
            }
        }
    }
}

// ---------------------------------------------------------------------------
// Launch
// ---------------------------------------------------------------------------
extern "C" void kernel_launch(void* const* d_in, const int* in_sizes, int n_in,
                              void* d_out, int out_size, void* d_ws, size_t ws_size,
                              hipStream_t stream)
{
    (void)in_sizes; (void)n_in; (void)out_size; (void)ws_size;

    const int*   x       = (const int*)  d_in[0];
    const float* emb     = (const float*)d_in[1];
    const float* head_w  = (const float*)d_in[2];
    const float* t0_proj = (const float*)d_in[3];
    const float* t0_out  = (const float*)d_in[4];
    const float* t1_proj = (const float*)d_in[5];
    const float* t1_out  = (const float*)d_in[6];
    float* out = (float*)d_out;

    // workspace carving (256B aligned)
    char* ws = (char*)d_ws;
    size_t off = 0;
    auto take = [&](size_t bytes) -> void* {
        void* p = ws + off;
        off += (bytes + 255) & ~(size_t)255;
        return p;
    };
    __bf16* pooled_b = (__bf16*)take((size_t)BATCH * EMBED * 2);
    __bf16* h0_b     = (__bf16*)take((size_t)BATCH * H0DIM * 2);
    __bf16* h1_b     = (__bf16*)take((size_t)BATCH * H1DIM * 2);
    __bf16* headW_b  = (__bf16*)take((size_t)HEAD_N * EMBED * 2);
    __bf16* t0W_b    = (__bf16*)take((size_t)T0_N * H0DIM * 2);
    __bf16* t1W_b    = (__bf16*)take((size_t)T1_N * H1DIM * 2);
    float*  head_lp  = (float*) take((size_t)BATCH * 2 * 4);

    // 1) pool + projections
    pool_proj_kernel<<<BATCH / 8, 256, 0, stream>>>(
        x, emb, t0_proj, t1_proj, pooled_b, h0_b, h1_b);

    // 2) weight conversion to bf16
    {
        const int n4h = HEAD_N * EMBED / 4;   // 400064
        cvt_bf16_kernel<<<(n4h + 255) / 256, 256, 0, stream>>>(head_w, headW_b, n4h);
        const int n40 = T0_N * H0DIM / 4;     // 200000
        cvt_bf16_kernel<<<(n40 + 255) / 256, 256, 0, stream>>>(t0_out, t0W_b, n40);
        const int n41 = T1_N * H1DIM / 4;     // 200000
        cvt_bf16_kernel<<<(n41 + 255) / 256, 256, 0, stream>>>(t1_out, t1W_b, n41);
    }

    // 3) head: GEMM(2048x12502,K=128) + log_softmax; writes cols [0,12500)
    //    and the two tail-gate log-probs.
    seg_softmax_kernel<4, 0><<<BATCH / 16, 256, 0, stream>>>(
        pooled_b, headW_b, head_lp, head_lp, out,
        HEAD_N, HEAD_N, CUT0, 0);

    // 4) tail0: GEMM(2048x12500,K=64) + log_softmax + head_lp[:,0]
    seg_softmax_kernel<2, 1><<<BATCH / 16, 256, 0, stream>>>(
        h0_b, t0W_b, head_lp, head_lp, out,
        T0_N, T0_N, T0_N, CUT0);

    // 5) tail1: GEMM(2048x25000,K=32) + log_softmax + head_lp[:,1]
    seg_softmax_kernel<1, 2><<<BATCH / 16, 256, 0, stream>>>(
        h1_b, t1W_b, head_lp, head_lp, out,
        T1_N, T1_N, T1_N, CUT0 + T0_N);
}
